// Decoder_80814104642079
// MI455X (gfx1250) — matrix operand order repaired and checked by reading, not compiled, not run on hardware
//
#include <hip/hip_runtime.h>

#define N_NODES 100000
#define N_EDGES 3200000

typedef __attribute__((ext_vector_type(16))) __bf16 v16bf;
typedef __attribute__((ext_vector_type(8)))  float  v8f;

// float -> bf16. On gfx1250 a plain fptrunc lowers to the native packed
// v_cvt bf16 hardware op (round-to-nearest-even), replacing the 4-VALU
// manual RNE sequence per element.
static __device__ __forceinline__ __bf16 f2bf(float f) {
#if defined(__AMDGCN__)
  return (__bf16)f;
#else
  unsigned u = __builtin_bit_cast(unsigned, f);
  unsigned r = u + 0x7FFFu + ((u >> 16) & 1u);
  unsigned short h = (unsigned short)(r >> 16);
  return __builtin_bit_cast(__bf16, h);
#endif
}

// C = A[rows x K] * W[K x ncols], fp32 in/out, bf16 WMMA with fp32 accumulate.
// One wave computes one 16x16 output tile; blockIdx.x = row tile, wave id = col tile.
// grid rows = 6250 * 16 == 100000 exactly -> EXEC is all-ones everywhere (WMMA req).
template <int K, int WAVES>
__global__ __launch_bounds__(WAVES * 32)
void gemm_bf16_wmma(const float* __restrict__ A, const float* __restrict__ W,
                    float* __restrict__ out, int ncols) {
  const int wave = threadIdx.x >> 5;      // column tile within block
  const int lane = threadIdx.x & 31;
  const int half = lane >> 4;             // 0: lanes 0-15, 1: lanes 16-31
  const int mr   = lane & 15;
  const long rowTile = blockIdx.x;
  const long arow = rowTile * 16 + mr;    // A row held by this lane
  const int  ncol = wave * 16 + mr;       // B/C/D column held by this lane

  v8f acc = {};
#pragma unroll
  for (int kt = 0; kt < K; kt += 32) {
    // A fragment (16-bit 16x32 layout): lane holds row `arow`,
    // elements 0..15 -> k = kt + half*16 + i   (one contiguous 16-float run)
    v16bf a;
    const float* ap = A + arow * K + kt + half * 16;
#pragma unroll
    for (int i = 0; i < 8; ++i) { a[i] = f2bf(ap[i]); a[8 + i] = f2bf(ap[8 + i]); }
    // B fragment (32x16): lane = column, element i -> k = kt + 16*half + i
    v16bf b;
    const float* wp = W + (long)(kt + 16 * half) * ncols + ncol;
#pragma unroll
    for (int i = 0; i < 16; ++i) b[i] = f2bf(wp[(long)i * ncols]);
    acc = __builtin_amdgcn_wmma_f32_16x16x32_bf16(
        /*neg_a=*/false, a, /*neg_b=*/false, b,
        /*c_mod=*/(short)0, acc, /*reuse_a=*/false, /*reuse_b=*/false);
  }
  // D (f32 16x16): VGPR r holds M = r + 8*half, N = lane&15
  float* op = out + (rowTile * 16 + 8 * half) * (long)ncols + ncol;
#pragma unroll
  for (int r = 0; r < 8; ++r) op[(long)r * ncols] = acc[r];
}

__global__ __launch_bounds__(256)
void zero_f32v4(float4* __restrict__ p, long n4) {
  long i = (long)blockIdx.x * blockDim.x + threadIdx.x;
  if (i < n4) p[i] = make_float4(0.f, 0.f, 0.f, 0.f);
}

// out[row[e], :] += val[e] * x[col[e], :]
// One thread handles one float4 chunk of one edge: float4 gather from x
// (L2-resident table) + 4 hardware global_atomic_add_f32 into out.
template <int F>
__global__ __launch_bounds__(256)
void spmm_scatter(const int* __restrict__ erow, const int* __restrict__ ecol,
                  const float* __restrict__ eval, const float* __restrict__ x,
                  float* __restrict__ out) {
  constexpr int CH = F / 4;                     // float4 chunks per feature row
  constexpr int SH = (CH == 16) ? 4 : 5;        // log2(CH), F in {64,128}
  const long t = (long)blockIdx.x * blockDim.x + threadIdx.x;
  if (t >= (long)N_EDGES * CH) return;
  const int e = (int)(t >> SH);
  const int c = (int)(t & (CH - 1));
  const float v = eval[e];
  const float4 xv = *(const float4*)(x + (long)ecol[e] * F + c * 4);
  float* o = out + (long)erow[e] * F + c * 4;
  unsafeAtomicAdd(o + 0, v * xv.x);
  unsafeAtomicAdd(o + 1, v * xv.y);
  unsafeAtomicAdd(o + 2, v * xv.z);
  unsafeAtomicAdd(o + 3, v * xv.w);
}

extern "C" void kernel_launch(void* const* d_in, const int* in_sizes, int n_in,
                              void* d_out, int out_size, void* d_ws, size_t ws_size,
                              hipStream_t stream) {
  const float* feat = (const float*)d_in[0];   // [N,32]
  const float* W1   = (const float*)d_in[1];   // [32,64]
  const float* W2   = (const float*)d_in[2];   // [64,128]
  const int*   erow = (const int*)d_in[3];     // [E]
  const int*   ecol = (const int*)d_in[4];     // [E]
  const float* eval = (const float*)d_in[5];   // [E]
  float* outp = (float*)d_out;                 // [N,128]

  // Scratch: bufA holds x1 [N,64] then x3 [N,128] (x1 dead before x3 written);
  // bufB holds x2 [N,64]. Total = N*192*4 = 76.8 MB.
  float* bufA = (float*)d_ws;
  float* bufB = bufA + (long)N_NODES * 128;

  const int rowTiles = N_NODES / 16;           // 6250, exact

  // x1 = feat @ W1
  gemm_bf16_wmma<32, 4><<<rowTiles, 128, 0, stream>>>(feat, W1, bufA, 64);

  // x2 = A @ x1 (scatter-add)
  {
    long n4 = (long)N_NODES * 64 / 4;
    zero_f32v4<<<(int)((n4 + 255) / 256), 256, 0, stream>>>((float4*)bufB, n4);
    long tt = (long)N_EDGES * 16;
    spmm_scatter<64><<<(int)((tt + 255) / 256), 256, 0, stream>>>(erow, ecol, eval, bufA, bufB);
  }

  // x3 = x2 @ W2
  gemm_bf16_wmma<64, 8><<<rowTiles, 256, 0, stream>>>(bufB, W2, bufA, 128);

  // out = A @ x3 (scatter-add)
  {
    long n4 = (long)N_NODES * 128 / 4;
    zero_f32v4<<<(int)((n4 + 255) / 256), 256, 0, stream>>>((float4*)outp, n4);
    long tt = (long)N_EDGES * 32;
    spmm_scatter<128><<<(int)((tt + 255) / 256), 256, 0, stream>>>(erow, ecol, eval, bufA, outp);
  }
}
